// Deformable_33397665693799
// MI455X (gfx1250) — matrix operand order repaired and checked by reading, not compiled, not run on hardware
//
#include <hip/hip_runtime.h>

// ---------------------------------------------------------------------------
// Deformable attention block for MI455X (gfx1250, wave32, WMMA).
// All GEMMs + attention use v_wmma_f32_16x16x32_f16 (f16 in, f32 accumulate).
// ---------------------------------------------------------------------------

typedef __attribute__((ext_vector_type(16))) _Float16 v16h;
typedef __attribute__((ext_vector_type(8)))  _Float16 v8h;
typedef __attribute__((ext_vector_type(8)))  float    v8f;

#define CC   512
#define HH   64
#define WW   64
#define HW   4096        // 64*64
#define HG   32
#define WG   32
#define PG   1024        // 32*32
#define BB   4
#define NHD  8
#define HD   64

static __device__ __forceinline__ v16h cat16(v8h lo, v8h hi) {
  return __builtin_shufflevector(lo, hi, 0,1,2,3,4,5,6,7,8,9,10,11,12,13,14,15);
}

static __device__ __forceinline__ float gelu_exact(float x) {
  return 0.5f * x * (1.0f + erff(x * 0.70710678118654752440f));
}

// ------------------------- weight transpose + f16 pack ----------------------
// wt[n*512 + k] = (f16) w[k*512 + n]   (so GEMM B-tiles are row-contiguous)
__global__ void pack_wT(const float* __restrict__ w, _Float16* __restrict__ wt) {
  int idx = blockIdx.x * 256 + threadIdx.x;      // 512*512 = 262144
  int n = idx >> 9, k = idx & 511;
  wt[idx] = (_Float16)w[k * CC + n];
}

// ------------------------------ LayerNorm -----------------------------------
// One block per row of 512 channels.  transposed=1: write (b,c,h,w) f16 layout
// (faithful to x_ln.transpose(0,3,1,2).reshape(b,hw,c) raw reinterpretation).
__global__ void __launch_bounds__(256)
ln_fwd(const float* __restrict__ in, const float* __restrict__ g,
       const float* __restrict__ bta, _Float16* __restrict__ outH, int transposed) {
  int row = blockIdx.x, t = threadIdx.x;
  const float* r = in + (size_t)row * CC;
  float v0 = r[t], v1 = r[t + 256];
  __shared__ float red[256];
  red[t] = v0 + v1;
  __syncthreads();
  for (int s = 128; s > 0; s >>= 1) { if (t < s) red[t] += red[t + s]; __syncthreads(); }
  float mean = red[0] * (1.0f / 512.0f);
  __syncthreads();
  float d0 = v0 - mean, d1 = v1 - mean;
  red[t] = d0 * d0 + d1 * d1;
  __syncthreads();
  for (int s = 128; s > 0; s >>= 1) { if (t < s) red[t] += red[t + s]; __syncthreads(); }
  float inv = rsqrtf(red[0] * (1.0f / 512.0f) + 1e-5f);
  float y0 = d0 * inv * g[t] + bta[t];
  float y1 = d1 * inv * g[t + 256] + bta[t + 256];
  if (transposed) {
    int b = row >> 12, pix = row & (HW - 1);
    outH[(((size_t)b * CC + t)       << 12) + pix] = (_Float16)y0;
    outH[(((size_t)b * CC + t + 256) << 12) + pix] = (_Float16)y1;
  } else {
    outH[(size_t)row * CC + t]       = (_Float16)y0;
    outH[(size_t)row * CC + t + 256] = (_Float16)y1;
  }
}

// ------------------------------ WMMA GEMM -----------------------------------
// C[M x 512] = A[M x 512] * W[512 x 512] + bias, A f16 row-major, Bt = W^T f16.
// Block: 256 threads = 8 waves as 4(M) x 2(N); tile 64(M) x 128(N), K step 32.
// epi: 0 = none, 1 = +res, 2 = gelu(.)+res
__global__ void __launch_bounds__(256)
gemm_wmma(const _Float16* __restrict__ A, const _Float16* __restrict__ Bt,
          const float* __restrict__ bias, const float* __restrict__ res,
          float* __restrict__ outF, _Float16* __restrict__ outH, int epi) {
  const int N = CC, K = CC;
  int t = threadIdx.x;
  int m0 = blockIdx.x * 64;
  int n0 = blockIdx.y * 128;
  __shared__ __attribute__((aligned(32))) _Float16 lA[64 * 32];
  __shared__ __attribute__((aligned(32))) _Float16 lB[128 * 32];
  int wid = t >> 5, lane = t & 31, lg = lane >> 4, lr = lane & 15;
  int wm = wid & 3, wn = wid >> 2;
  int rowA = wm * 16 + lr;

  v8f acc[4];
#pragma unroll
  for (int cn = 0; cn < 4; ++cn) {
    float bv = bias[n0 + wn * 64 + cn * 16 + lr];
#pragma unroll
    for (int g = 0; g < 8; ++g) acc[cn][g] = bv;
  }

  int arow = t >> 2, akofs = (t & 3) << 3;   // 64 rows x 8 halves
  int brow = t >> 1, bkofs = (t & 1) << 4;   // 128 rows x 16 halves

  for (int k0 = 0; k0 < K; k0 += 32) {
    __syncthreads();
    *(v8h*)&lA[arow * 32 + akofs] =
        *(const v8h*)&A[(size_t)(m0 + arow) * K + k0 + akofs];
    *(v8h*)&lB[brow * 32 + bkofs] =
        *(const v8h*)&Bt[(size_t)(n0 + brow) * K + k0 + bkofs];
    *(v8h*)&lB[brow * 32 + bkofs + 8] =
        *(const v8h*)&Bt[(size_t)(n0 + brow) * K + k0 + bkofs + 8];
    if (k0 + 32 < K)
      __builtin_prefetch(&A[(size_t)(m0 + arow) * K + k0 + 32 + akofs], 0, 0);
    __syncthreads();

    v8h alo = *(const v8h*)&lA[rowA * 32 + 8 * lg];
    v8h ahi = *(const v8h*)&lA[rowA * 32 + 16 + 8 * lg];
    v16h af = cat16(alo, ahi);
#pragma unroll
    for (int cn = 0; cn < 4; ++cn) {
      int nn = wn * 64 + cn * 16 + lr;
      v8h blo = *(const v8h*)&lB[nn * 32 + 8 * lg];
      v8h bhi = *(const v8h*)&lB[nn * 32 + 16 + 8 * lg];
      v16h bf = cat16(blo, bhi);
      acc[cn] = __builtin_amdgcn_wmma_f32_16x16x32_f16(
          false, af, false, bf, (short)0, acc[cn], false, false);
    }
  }

#pragma unroll
  for (int cn = 0; cn < 4; ++cn) {
    int col = n0 + wn * 64 + cn * 16 + lr;
#pragma unroll
    for (int g = 0; g < 8; ++g) {
      int row = m0 + wm * 16 + (lg << 3) + g;
      float v = acc[cn][g];
      size_t o = (size_t)row * N + col;
      if (epi == 1)      v += res[o];
      else if (epi == 2) v = gelu_exact(v) + res[o];
      if (outF) outF[o] = v;
      if (outH) outH[o] = (_Float16)v;
    }
  }
}

// --------------------- depthwise conv 7x7 stride 2 + GELU -------------------
// q viewed as (b,512,64,64) flat; out (b,512,32,32) flat.
__global__ void __launch_bounds__(256)
dwconv_kernel(const float* __restrict__ q, const float* __restrict__ w,
              const float* __restrict__ bias, float* __restrict__ o) {
  int idx = blockIdx.x * 256 + threadIdx.x;   // 4*512*32*32 = 2097152
  int ox = idx & 31, oy = (idx >> 5) & 31, c = (idx >> 10) & 511, b = idx >> 19;
  const float* wp = w + c * 49;
  const float* qp = q + (((size_t)b * CC + c) << 12);
  float acc = bias[c];
#pragma unroll
  for (int ky = 0; ky < 7; ++ky) {
    int iy = oy * 2 - 3 + ky;
    if (iy < 0 || iy >= HH) continue;
#pragma unroll
    for (int kx = 0; kx < 7; ++kx) {
      int ix = ox * 2 - 3 + kx;
      if (ix < 0 || ix >= WW) continue;
      acc += qp[iy * WW + ix] * wp[ky * 7 + kx];
    }
  }
  o[idx] = gelu_exact(acc);
}

// ------------------------- pointwise conv to 2 channels ---------------------
__global__ void __launch_bounds__(256)
pw_kernel(const float* __restrict__ o, const float* __restrict__ pw,
          float* __restrict__ offs) {
  int p = blockIdx.x;             // 4096 pixels
  int b = p >> 10, rem = p & (PG - 1), t = threadIdx.x;
  float a0 = 0.f, a1 = 0.f;
  for (int c = t; c < CC; c += 256) {
    float ov = o[(((size_t)b * CC + c) << 10) + rem];
    a0 += pw[c] * ov;
    a1 += pw[CC + c] * ov;
  }
  __shared__ float r0[256], r1[256];
  r0[t] = a0; r1[t] = a1;
  __syncthreads();
  for (int s = 128; s > 0; s >>= 1) {
    if (t < s) { r0[t] += r0[t + s]; r1[t] += r1[t + s]; }
    __syncthreads();
  }
  if (t == 0) {
    offs[(size_t)b * 2 * PG + rem]      = r0[0];
    offs[(size_t)b * 2 * PG + PG + rem] = r1[0];
  }
}

// ---------------- tanh offsets + grid build + bilinear sample ---------------
// Faithful to reference: refs from meshgrid(ij), buggy *(h/R)/2 scaling,
// channel flip before sampling, zeros padding, align_corners=True.
__global__ void __launch_bounds__(256)
grid_sample_kernel(const float* __restrict__ x, const float* __restrict__ offs,
                   _Float16* __restrict__ interp) {
  int p = blockIdx.x;             // (b, pp, qq)
  int b = p >> 10, rem = p & (PG - 1);
  int pp = rem >> 5, qq = rem & 31;
  float o0 = offs[(size_t)b * 2 * PG + rem];
  float o1 = offs[(size_t)b * 2 * PG + PG + rem];
  float refx = (pp + 0.5f) * (2.0f / 32.0f) - 1.0f;
  float refy = (qq + 0.5f) * (2.0f / 32.0f) - 1.0f;
  float d0 = (refx + tanhf(o0) + 1.0f) * 16.0f;   // channel 0
  float d1 = (refy + tanhf(o1) + 1.0f) * 16.0f;   // channel 1
  // grid[...,0]=d1 (x), grid[...,1]=d0 (y)
  float xs = (d1 + 1.0f) * 0.5f * 63.0f;
  float ys = (d0 + 1.0f) * 0.5f * 63.0f;
  float x0f = floorf(xs), y0f = floorf(ys);
  float wx = xs - x0f, wy = ys - y0f;
  int x0 = (int)x0f, y0 = (int)y0f;
  int x1 = x0 + 1, y1 = y0 + 1;
  float w00 = (1.f - wx) * (1.f - wy), w10 = wx * (1.f - wy);
  float w01 = (1.f - wx) * wy,         w11 = wx * wy;
  w00 = (x0 >= 0 && x0 < WW && y0 >= 0 && y0 < HH) ? w00 : 0.f;
  w10 = (x1 >= 0 && x1 < WW && y0 >= 0 && y0 < HH) ? w10 : 0.f;
  w01 = (x0 >= 0 && x0 < WW && y1 >= 0 && y1 < HH) ? w01 : 0.f;
  w11 = (x1 >= 0 && x1 < WW && y1 >= 0 && y1 < HH) ? w11 : 0.f;
  int xc0 = min(max(x0, 0), WW - 1), xc1 = min(max(x1, 0), WW - 1);
  int yc0 = min(max(y0, 0), HH - 1), yc1 = min(max(y1, 0), HH - 1);
  size_t b00 = (((size_t)b * HH + yc0) * WW + xc0) * CC;
  size_t b10 = (((size_t)b * HH + yc0) * WW + xc1) * CC;
  size_t b01 = (((size_t)b * HH + yc1) * WW + xc0) * CC;
  size_t b11 = (((size_t)b * HH + yc1) * WW + xc1) * CC;
  int t = threadIdx.x;
#pragma unroll
  for (int i = 0; i < 2; ++i) {
    int c = t + i * 256;
    float acc = w00 * x[b00 + c] + w10 * x[b10 + c]
              + w01 * x[b01 + c] + w11 * x[b11 + c];
    interp[(((size_t)b * CC + c) << 10) + rem] = (_Float16)acc;   // (b,c,32,32)
  }
}

// --------------------------- flash attention --------------------------------
// grid (64 qtiles, 8 heads, 4 batch); 128 threads = 4 waves, 16 q-rows/wave.
// S = Q K^T (no scale, faithful), online softmax, O = P V.  All WMMA f16.
__global__ void __launch_bounds__(128)
attn_kernel(const _Float16* __restrict__ qh, const _Float16* __restrict__ kh,
            const _Float16* __restrict__ vh, _Float16* __restrict__ att) {
  int t = threadIdx.x;
  int wid = t >> 5, lane = t & 31, lg = lane >> 4, lr = lane & 15;
  int bi = blockIdx.z, hd = blockIdx.y;
  int q0 = blockIdx.x * 64 + wid * 16;

  __shared__ __attribute__((aligned(32))) _Float16 lK[32 * 64];   // [key][d]
  __shared__ __attribute__((aligned(32))) _Float16 lVt[64 * 32];  // [d][key]
  __shared__ __attribute__((aligned(32))) _Float16 lP[4][16 * 32];

  // Q fragments (16 rows x 64 d -> two 16x32 A-fragments)
  size_t qbase = ((size_t)(bi * HW + q0 + lr)) * CC + hd * HD;
  v16h qf[2];
#pragma unroll
  for (int ch = 0; ch < 2; ++ch) {
    v8h lo = *(const v8h*)&qh[qbase + ch * 32 + 8 * lg];
    v8h hi = *(const v8h*)&qh[qbase + ch * 32 + 16 + 8 * lg];
    qf[ch] = cat16(lo, hi);
  }

  float mrun[8], lrun[8];
#pragma unroll
  for (int g = 0; g < 8; ++g) { mrun[g] = -1e30f; lrun[g] = 0.f; }
  v8f o[4];
#pragma unroll
  for (int cn = 0; cn < 4; ++cn)
#pragma unroll
    for (int g = 0; g < 8; ++g) o[cn][g] = 0.f;

  int sj = t >> 2, sdo = (t & 3) << 4;   // staging: 32 keys x 16 halves

  for (int kt = 0; kt < PG; kt += 32) {
    __syncthreads();
    size_t kb = ((size_t)(bi * PG + kt + sj)) * CC + hd * HD + sdo;
    v8h k0v = *(const v8h*)&kh[kb];
    v8h k1v = *(const v8h*)&kh[kb + 8];
    *(v8h*)&lK[sj * 64 + sdo]     = k0v;
    *(v8h*)&lK[sj * 64 + sdo + 8] = k1v;
    v8h v0v = *(const v8h*)&vh[kb];
    v8h v1v = *(const v8h*)&vh[kb + 8];
#pragma unroll
    for (int i = 0; i < 8; ++i) {
      lVt[(sdo + i)     * 32 + sj] = v0v[i];
      lVt[(sdo + 8 + i) * 32 + sj] = v1v[i];
    }
    __syncthreads();

    // S tiles: columns [0..15] and [16..31] of this 32-key chunk
    v8f s[2];
#pragma unroll
    for (int sc = 0; sc < 2; ++sc) {
#pragma unroll
      for (int g = 0; g < 8; ++g) s[sc][g] = 0.f;
      int j = sc * 16 + lr;
#pragma unroll
      for (int dc = 0; dc < 2; ++dc) {
        v8h blo = *(const v8h*)&lK[j * 64 + dc * 32 + 8 * lg];
        v8h bhi = *(const v8h*)&lK[j * 64 + dc * 32 + 16 + 8 * lg];
        v16h bf = cat16(blo, bhi);
        s[sc] = __builtin_amdgcn_wmma_f32_16x16x32_f16(
            false, qf[dc], false, bf, (short)0, s[sc], false, false);
      }
    }

    // online softmax (row stats live in the 16-lane group holding that row)
    float fac[8];
#pragma unroll
    for (int g = 0; g < 8; ++g) {
      float vm = fmaxf(s[0][g], s[1][g]);
      vm = fmaxf(vm, __shfl_xor(vm, 1, 32));
      vm = fmaxf(vm, __shfl_xor(vm, 2, 32));
      vm = fmaxf(vm, __shfl_xor(vm, 4, 32));
      vm = fmaxf(vm, __shfl_xor(vm, 8, 32));
      float mn = fmaxf(mrun[g], vm);
      fac[g] = expf(mrun[g] - mn);
      float p0 = expf(s[0][g] - mn);
      float p1 = expf(s[1][g] - mn);
      s[0][g] = p0; s[1][g] = p1;
      float rs = p0 + p1;
      rs += __shfl_xor(rs, 1, 32);
      rs += __shfl_xor(rs, 2, 32);
      rs += __shfl_xor(rs, 4, 32);
      rs += __shfl_xor(rs, 8, 32);
      lrun[g] = lrun[g] * fac[g] + rs;
      mrun[g] = mn;
    }
#pragma unroll
    for (int cn = 0; cn < 4; ++cn)
#pragma unroll
      for (int g = 0; g < 8; ++g) o[cn][g] *= fac[g];

    // P (C/D layout) -> per-wave LDS -> A-fragment layout (DS is in-order)
#pragma unroll
    for (int g = 0; g < 8; ++g) {
      int r = (lg << 3) + g;
      lP[wid][r * 32 + lr]      = (_Float16)s[0][g];
      lP[wid][r * 32 + 16 + lr] = (_Float16)s[1][g];
    }
    v8h plo = *(const v8h*)&lP[wid][lr * 32 + 8 * lg];
    v8h phi = *(const v8h*)&lP[wid][lr * 32 + 16 + 8 * lg];
    v16h pf = cat16(plo, phi);

    // O += P * V  (K=32 keys, N=64 d split into 4 chunks)
#pragma unroll
    for (int cn = 0; cn < 4; ++cn) {
      int d = cn * 16 + lr;
      v8h blo = *(const v8h*)&lVt[d * 32 + 8 * lg];
      v8h bhi = *(const v8h*)&lVt[d * 32 + 16 + 8 * lg];
      v16h bf = cat16(blo, bhi);
      o[cn] = __builtin_amdgcn_wmma_f32_16x16x32_f16(
          false, pf, false, bf, (short)0, o[cn], false, false);
    }
  }

#pragma unroll
  for (int cn = 0; cn < 4; ++cn)
#pragma unroll
    for (int g = 0; g < 8; ++g) {
      int row = q0 + (lg << 3) + g;
      float val = o[cn][g] / lrun[g];
      att[((size_t)(bi * HW + row)) * CC + hd * HD + cn * 16 + lr] = (_Float16)val;
    }
}

// ---------------------------------------------------------------------------
extern "C" void kernel_launch(void* const* d_in, const int* in_sizes, int n_in,
                              void* d_out, int out_size, void* d_ws, size_t ws_size,
                              hipStream_t stream) {
  (void)in_sizes; (void)n_in; (void)out_size; (void)ws_size;
  const float* x     = (const float*)d_in[0];
  const float* ln_g  = (const float*)d_in[1];
  const float* ln_b  = (const float*)d_in[2];
  const float* wq    = (const float*)d_in[3];
  const float* bq    = (const float*)d_in[4];
  const float* wv    = (const float*)d_in[5];
  const float* bv    = (const float*)d_in[6];
  const float* dw_w  = (const float*)d_in[7];
  const float* dw_b  = (const float*)d_in[8];
  const float* pw_w  = (const float*)d_in[9];
  const float* mq_w  = (const float*)d_in[10];
  const float* mq_b  = (const float*)d_in[11];
  const float* mk_w  = (const float*)d_in[12];
  const float* mk_b  = (const float*)d_in[13];
  const float* mv_w  = (const float*)d_in[14];
  const float* mv_b  = (const float*)d_in[15];
  const float* mo_w  = (const float*)d_in[16];
  const float* mo_b  = (const float*)d_in[17];
  const float* mlp_w = (const float*)d_in[18];
  const float* mlp_b = (const float*)d_in[19];
  float* out = (float*)d_out;

  char* ws = (char*)d_ws;
  size_t off = 0;
  auto alloc = [&](size_t bytes) -> char* {
    char* p = ws + off;
    off = (off + bytes + 255) & ~(size_t)255;
    return p;
  };
  const size_t WB = (size_t)CC * CC * sizeof(_Float16);
  _Float16* wqT   = (_Float16*)alloc(WB);
  _Float16* wvT   = (_Float16*)alloc(WB);
  _Float16* mqT   = (_Float16*)alloc(WB);
  _Float16* mkT   = (_Float16*)alloc(WB);
  _Float16* mvT   = (_Float16*)alloc(WB);
  _Float16* moT   = (_Float16*)alloc(WB);
  _Float16* mlpT  = (_Float16*)alloc(WB);
  _Float16* xlnH  = (_Float16*)alloc((size_t)BB * HW * CC * 2);
  float*    qF    = (float*)   alloc((size_t)BB * HW * CC * 4);
  _Float16* qH    = (_Float16*)alloc((size_t)BB * HW * CC * 2);
  float*    oF    = (float*)   alloc((size_t)BB * CC * PG * 4);
  float*    offsF = (float*)   alloc((size_t)BB * 2 * PG * 4);
  _Float16* intH  = (_Float16*)alloc((size_t)BB * CC * PG * 2);
  _Float16* vpreH = (_Float16*)alloc((size_t)BB * PG * CC * 2);
  _Float16* qhH   = (_Float16*)alloc((size_t)BB * HW * CC * 2);
  _Float16* khH   = (_Float16*)alloc((size_t)BB * PG * CC * 2);
  _Float16* vhH   = (_Float16*)alloc((size_t)BB * PG * CC * 2);
  _Float16* attH  = (_Float16*)alloc((size_t)BB * HW * CC * 2);
  float*    zF    = (float*)   alloc((size_t)BB * HW * CC * 4);
  _Float16* zlnH  = (_Float16*)alloc((size_t)BB * HW * CC * 2);

  hipStream_t s = stream;
  // weight packs (transpose + f16)
  pack_wT<<<1024, 256, 0, s>>>(wq,    wqT);
  pack_wT<<<1024, 256, 0, s>>>(wv,    wvT);
  pack_wT<<<1024, 256, 0, s>>>(mq_w,  mqT);
  pack_wT<<<1024, 256, 0, s>>>(mk_w,  mkT);
  pack_wT<<<1024, 256, 0, s>>>(mv_w,  mvT);
  pack_wT<<<1024, 256, 0, s>>>(mo_w,  moT);
  pack_wT<<<1024, 256, 0, s>>>(mlp_w, mlpT);

  // LN1 -> (b,c,h,w) f16 (faithful to transpose+raw-reshape)
  ln_fwd<<<BB * HW, 256, 0, s>>>(x, ln_g, ln_b, xlnH, 1);
  // q = x_flat @ wq + bq    (M=16384)
  gemm_wmma<<<dim3(256, 4), 256, 0, s>>>(xlnH, wqT, bq, nullptr, qF, qH, 0);
  // depthwise conv + gelu, pointwise conv, deform grid + bilinear sample
  dwconv_kernel<<<8192, 256, 0, s>>>(qF, dw_w, dw_b, oF);
  pw_kernel<<<BB * PG, 256, 0, s>>>(oF, pw_w, offsF);
  grid_sample_kernel<<<BB * PG, 256, 0, s>>>(x, offsF, intH);
  // v_pre = interp_flat @ wv + bv  (k uses the same projection, faithful)
  gemm_wmma<<<dim3(64, 4), 256, 0, s>>>(intH, wvT, bv, nullptr, nullptr, vpreH, 0);
  // qh / kh / vh projections
  gemm_wmma<<<dim3(256, 4), 256, 0, s>>>(qH,    mqT, mq_b, nullptr, nullptr, qhH, 0);
  gemm_wmma<<<dim3(64, 4),  256, 0, s>>>(vpreH, mkT, mk_b, nullptr, nullptr, khH, 0);
  gemm_wmma<<<dim3(64, 4),  256, 0, s>>>(vpreH, mvT, mv_b, nullptr, nullptr, vhH, 0);
  // flash attention
  attn_kernel<<<dim3(64, NHD, BB), 128, 0, s>>>(qhH, khH, vhH, attH);
  // z = att @ mo_w + mo_b + x
  gemm_wmma<<<dim3(256, 4), 256, 0, s>>>(attH, moT, mo_b, x, zF, nullptr, 1);
  // LN2 (shared params, faithful)
  ln_fwd<<<BB * HW, 256, 0, s>>>(zF, ln_g, ln_b, zlnH, 0);
  // out = gelu(z_ln @ mlp_w + mlp_b) + z   (raw reshape to (b,w,h,c) is flat)
  gemm_wmma<<<dim3(256, 4), 256, 0, s>>>(zlnH, mlpT, mlp_b, zF, out, nullptr, 2);
}
